// VecKM_82970178224386
// MI455X (gfx1250) — matrix operand order repaired and checked by reading, not compiled, not run on hardware
//
#include <hip/hip_runtime.h>
#include <hip/hip_bf16.h>

typedef _Float16 half16 __attribute__((ext_vector_type(16)));
typedef _Float16 half8  __attribute__((ext_vector_type(8)));
typedef float    f32x8  __attribute__((ext_vector_type(8)));

#define N1 8192
#define N2 8192
#define DD 256
#define BETA2 40.5f

// ---------------------------------------------------------------------------
// Kernel 1: Bt[c][j] = c<256 ? cos(x2_j . A[:,c]) : sin(x2_j . A[:,c-256])
// Stored K-major (512 x 8192, f16) so WMMA B-fragments are contiguous loads.
// grid (32, 512) x 256 threads
// ---------------------------------------------------------------------------
__global__ __launch_bounds__(256) void veckm_feat(
    const float* __restrict__ x2, const float* __restrict__ A,
    _Float16* __restrict__ Bt) {
  int j = blockIdx.x * 256 + threadIdx.x;  // 0..8191
  int c = blockIdx.y;                      // 0..511
  float px = x2[j * 3 + 0], py = x2[j * 3 + 1], pz = x2[j * 3 + 2];
  int cc = c & 255;
  float th = px * A[cc] + py * A[DD + cc] + pz * A[2 * DD + cc];
  float v = (c < DD) ? __cosf(th) : __sinf(th);
  Bt[(size_t)c * N2 + j] = (_Float16)v;
}

// ---------------------------------------------------------------------------
// Kernel 2: G = J @ B fused.  One block = 16-row x 512-col output stripe.
// J tile (16x32) computed on the fly -> LDS f16 (ping-pong, 1 barrier/iter),
// consumed as WMMA A fragments; B fragments loaded contiguously from Bt.
// Writes f32 D fragments straight into d_out with (n,256,2) interleave.
// grid 512 x 256 threads (8 waves)
// ---------------------------------------------------------------------------
__global__ __launch_bounds__(256) void veckm_gemm(
    const float* __restrict__ x1, const float* __restrict__ x2,
    const _Float16* __restrict__ Bt, float* __restrict__ out) {
  __shared__ alignas(16) _Float16 Jt[2][16][32];
  __shared__ float X1[16][4];

  const int t  = threadIdx.x;
  const int i0 = blockIdx.x * 16;

  if (t < 16) {
    X1[t][0] = x1[(i0 + t) * 3 + 0];
    X1[t][1] = x1[(i0 + t) * 3 + 1];
    X1[t][2] = x1[(i0 + t) * 3 + 2];
  }

  const int wid = t >> 5;          // wave id 0..7 (wave32)
  const int ln  = t & 31;          // lane
  const int n0  = wid * 64;        // this wave's column base (4 tiles of 16)
  // A-fragment lane mapping (16-bit A 16x32): row = ln%16, K base 0/16
  const int mA  = ln & 15;
  const int kbA = (ln >> 4) * 16;
  // B-fragment lane mapping (16-bit B 32x16): col = ln%16, K base 0/16
  const int cB  = ln & 15;
  const int kbB = (ln >> 4) * 16;

  f32x8 acc0 = {}, acc1 = {}, acc2 = {}, acc3 = {};

  __syncthreads();

  // first J tile (k0 = 0) into buffer 0
#pragma unroll
  for (int q = 0; q < 2; ++q) {
    int p = t + q * 256;            // 0..511 over 16x32 tile
    int m = p >> 5, k = p & 31;
    int j = k;
    float dx = X1[m][0] - x2[j * 3 + 0];
    float dy = X1[m][1] - x2[j * 3 + 1];
    float dz = X1[m][2] - x2[j * 3 + 2];
    Jt[0][m][k] = (_Float16)__expf(-BETA2 * (dx * dx + dy * dy + dz * dz));
  }
  __syncthreads();

  for (int it = 0; it < N2 / 32; ++it) {
    const int k0  = it * 32;
    const int buf = it & 1;

    // --- A fragment from LDS J tile (contiguous K order k = 16h + s, same as B) ---
    half8 alo = *(const half8*)&Jt[buf][mA][kbA];        // K = kb..kb+7
    half8 ahi = *(const half8*)&Jt[buf][mA][kbA + 8];    // K = kb+8..kb+15
    half16 a;
#pragma unroll
    for (int e = 0; e < 8; ++e) { a[e] = alo[e]; a[e + 8] = ahi[e]; }

    // --- B fragments: 16 contiguous K values per lane from K-major Bt ---
    const _Float16* bbase = Bt + (size_t)(n0 + cB) * N2 + k0 + kbB;
    half16 b0 = *(const half16*)(bbase + (size_t)0 * 16 * N2);
    half16 b1 = *(const half16*)(bbase + (size_t)1 * 16 * N2);
    half16 b2 = *(const half16*)(bbase + (size_t)2 * 16 * N2);
    half16 b3 = *(const half16*)(bbase + (size_t)3 * 16 * N2);

    acc0 = __builtin_amdgcn_wmma_f32_16x16x32_f16(false, a, false, b0,
                                                  (short)0, acc0, false, false);
    acc1 = __builtin_amdgcn_wmma_f32_16x16x32_f16(false, a, false, b1,
                                                  (short)0, acc1, false, false);
    acc2 = __builtin_amdgcn_wmma_f32_16x16x32_f16(false, a, false, b2,
                                                  (short)0, acc2, false, false);
    acc3 = __builtin_amdgcn_wmma_f32_16x16x32_f16(false, a, false, b3,
                                                  (short)0, acc3, false, false);

    // --- compute next J tile into the other buffer (uniform branch) ---
    if (it + 1 < N2 / 32) {
      const int k0n = k0 + 32;
      const int bn  = buf ^ 1;
#pragma unroll
      for (int q = 0; q < 2; ++q) {
        int p = t + q * 256;
        int m = p >> 5, k = p & 31;
        int j = k0n + k;
        float dx = X1[m][0] - x2[j * 3 + 0];
        float dy = X1[m][1] - x2[j * 3 + 1];
        float dz = X1[m][2] - x2[j * 3 + 2];
        Jt[bn][m][k] = (_Float16)__expf(-BETA2 * (dx * dx + dy * dy + dz * dz));
      }
    }
    __syncthreads();
  }

  // --- store D fragments: VGPR r -> row r (lanes 0-15) / r+8 (lanes 16-31)
  const int mD = (ln >> 4) * 8;
#pragma unroll
  for (int r = 0; r < 8; ++r) {
    int i = i0 + mD + r;
    int nA = n0 + 0 * 16 + (ln & 15);
    int nB = n0 + 1 * 16 + (ln & 15);
    int nC = n0 + 2 * 16 + (ln & 15);
    int nE = n0 + 3 * 16 + (ln & 15);
    out[(size_t)i * 512 + (nA & 255) * 2 + (nA >> 8)] = acc0[r];
    out[(size_t)i * 512 + (nB & 255) * 2 + (nB >> 8)] = acc1[r];
    out[(size_t)i * 512 + (nC & 255) * 2 + (nC >> 8)] = acc2[r];
    out[(size_t)i * 512 + (nE & 255) * 2 + (nE >> 8)] = acc3[r];
  }
}

// ---------------------------------------------------------------------------
// Kernel 3: in-place epilogue on d_out.
// G *= conj(e^{i x1.A}); G *= 16/||G||; G += e^{i x1.P}
// grid 8192 x 256 threads (one block per row, one thread per feature)
// ---------------------------------------------------------------------------
__global__ __launch_bounds__(256) void veckm_final(
    const float* __restrict__ x1, const float* __restrict__ A,
    const float* __restrict__ P, float* __restrict__ out) {
  int i = blockIdx.x, c = threadIdx.x;
  float px = x1[i * 3 + 0], py = x1[i * 3 + 1], pz = x1[i * 3 + 2];

  float th = px * A[c] + py * A[DD + c] + pz * A[2 * DD + c];
  float co = __cosf(th), si = __sinf(th);

  float gr = out[(size_t)i * 512 + 2 * c];
  float gi = out[(size_t)i * 512 + 2 * c + 1];
  // (gr + i gi) * (co - i si)
  float re = gr * co + gi * si;
  float im = gi * co - gr * si;

  __shared__ float red[256];
  red[c] = re * re + im * im;
  __syncthreads();
  for (int s = 128; s > 0; s >>= 1) {
    if (c < s) red[c] += red[c + s];
    __syncthreads();
  }
  float scale = 16.0f * rsqrtf(red[0]);  // sqrt(256) / ||G||

  float ph = px * P[c] + py * P[DD + c] + pz * P[2 * DD + c];
  out[(size_t)i * 512 + 2 * c]     = re * scale + __cosf(ph);
  out[(size_t)i * 512 + 2 * c + 1] = im * scale + __sinf(ph);
}

// ---------------------------------------------------------------------------
extern "C" void kernel_launch(void* const* d_in, const int* in_sizes, int n_in,
                              void* d_out, int out_size, void* d_ws,
                              size_t ws_size, hipStream_t stream) {
  const float* p1 = (const float*)d_in[0];  // (8192,3)
  const float* p2 = (const float*)d_in[1];  // (8192,3)
  const float* A  = (const float*)d_in[2];  // (3,256)
  const float* P  = (const float*)d_in[3];  // (3,256)
  float* out      = (float*)d_out;          // (8192,256,2)
  _Float16* Bt    = (_Float16*)d_ws;        // (512,8192) f16, 8 MB

  veckm_feat<<<dim3(32, 512), 256, 0, stream>>>(p2, A, Bt);
  veckm_gemm<<<N1 / 16, 256, 0, stream>>>(p1, p2, Bt, out);
  veckm_final<<<N1, 256, 0, stream>>>(p1, A, P, out);
}
